// SpatialContext_63428077027658
// MI455X (gfx1250) — matrix operand order repaired and checked by reading, not compiled, not run on hardware
//
#include <hip/hip_runtime.h>

// ---------------------------------------------------------------------------
// Types for CDNA5 WMMA (gfx1250, wave32)
// ---------------------------------------------------------------------------
typedef __attribute__((ext_vector_type(16))) __bf16 v16bf;
typedef __attribute__((ext_vector_type(8)))  __bf16 bf16x8;
typedef __attribute__((ext_vector_type(8)))  float  v8f;

static __device__ __forceinline__ __bf16 f2bf(float f) {
  unsigned u = __builtin_bit_cast(unsigned, f);
  unsigned r = u + 0x7FFFu + ((u >> 16) & 1u);   // round-to-nearest-even
  unsigned short h = (unsigned short)(r >> 16);
  return __builtin_bit_cast(__bf16, h);
}

static __device__ __forceinline__ v8f wmma_bf16(v16bf a, v16bf b, v8f c) {
  // D(16x16 f32) = A(16x32 bf16) * B(32x16 bf16) + C
  return __builtin_amdgcn_wmma_f32_16x16x32_bf16(
      /*neg_a=*/false, a, /*neg_b=*/false, b,
      /*c_mod=*/(short)0, c, /*reuse_a=*/false, /*reuse_b=*/false);
}

static __device__ __forceinline__ v16bf pack_a(bf16x8 x, bf16x8 y) {
  v16bf a;
#pragma unroll
  for (int i = 0; i < 8; ++i) { a[i] = x[i]; a[i + 8] = y[i]; }
  return a;
}

static __device__ __forceinline__ float sigm(float x) {
  return 1.0f / (1.0f + __expf(-x));
}

// ---------------------------------------------------------------------------
// Problem constants
// ---------------------------------------------------------------------------
#define NBOX   2048
#define MM     4096      // MASK*MASK
#define NHOI   4096      // B_IMG * T_SEQ
#define BIMG   16
#define TSEQ   256
#define DHID   128
#define G4     512       // 4 * D_RNN

// workspace byte offsets
#define WS_MASKS 0u                      // 2048*4096 bf16  = 16 MB
#define WS_WCAT  16777216u               // 256*4096  bf16  =  2 MB
#define WS_P     18874368u               // 2048*256  f32   =  2 MB
#define WS_FEATS 20971520u               // 4096*128  bf16  =  1 MB
#define WS_WIH   22020096u               // 1024*128  bf16  = 256 KB
#define WS_BIAS  22282240u               // 1024      f32   =   4 KB
#define WS_WHH   22286336u               // 2*512*128 bf16  = 256 KB
#define WS_XG    22548480u               // 4096*1024 f32   = 16 MB

// ---------------------------------------------------------------------------
// Prep kernels (f32 -> bf16, weight concatenation, bias fold)
// ---------------------------------------------------------------------------
__global__ void k_cvt_bf16(const float* __restrict__ s, __bf16* __restrict__ d, int n) {
  for (int i = blockIdx.x * blockDim.x + threadIdx.x; i < n; i += gridDim.x * blockDim.x)
    d[i] = f2bf(s[i]);
}

// Wcat[256][4096]: rows 0..127 = fc_w[:, :4096], rows 128..255 = fc_w[:, 4096:]
__global__ void k_build_wcat(const float* __restrict__ fcw, __bf16* __restrict__ d) {
  int n = 256 * MM;
  for (int i = blockIdx.x * blockDim.x + threadIdx.x; i < n; i += gridDim.x * blockDim.x) {
    int r = i >> 12, k = i & (MM - 1);
    d[i] = f2bf(fcw[(r & 127) * (2 * MM) + (r >> 7) * MM + k]);
  }
}

// WihCat[1024][128]: rows 0..511 = w_ih_f, rows 512..1023 = w_ih_b
__global__ void k_build_wih(const float* __restrict__ wf, const float* __restrict__ wb,
                            __bf16* __restrict__ d) {
  int n = 1024 * DHID;
  for (int i = blockIdx.x * blockDim.x + threadIdx.x; i < n; i += gridDim.x * blockDim.x) {
    int r = i >> 7, k = i & 127;
    d[i] = f2bf(r < G4 ? wf[r * DHID + k] : wb[(r - G4) * DHID + k]);
  }
}

__global__ void k_build_bias(const float* __restrict__ bif, const float* __restrict__ bhf,
                             const float* __restrict__ bib, const float* __restrict__ bhb,
                             float* __restrict__ d) {
  int i = blockIdx.x * blockDim.x + threadIdx.x;
  if (i < 1024)
    d[i] = (i < G4) ? (bif[i] + bhf[i]) : (bib[i - G4] + bhb[i - G4]);
}

// ---------------------------------------------------------------------------
// GEMM 1: P[2048x256] = masks_bf16[2048x4096] @ Wcat^T  (per-box features)
// one wave => 16x32 output tile (2 N-tiles, shared A fragment)
// K-loop software-pipelined: next chunk's fragments load under current wmma.
// ---------------------------------------------------------------------------
__global__ void k_gemm_box(const __bf16* __restrict__ Mb, const __bf16* __restrict__ Wc,
                           float* __restrict__ P) {
  const int wave = blockIdx.x * (blockDim.x >> 5) + (threadIdx.x >> 5);
  const int lane = threadIdx.x & 31;
  const int mtile = wave >> 3;        // 0..127
  const int ng    = wave & 7;         // 0..7  (32 cols each)
  const int m0 = mtile * 16, n0 = ng * 32;
  const int row = lane & 15, hi = lane >> 4;

  const __bf16* arow  = Mb + (m0 + row) * MM + hi * 16;
  const __bf16* b0row = Wc + (n0 + row) * MM + hi * 16;
  const __bf16* b1row = Wc + (n0 + 16 + row) * MM + hi * 16;

  v8f c0 = {}; v8f c1 = {};
  // prologue: fragment set for kk = 0
  v16bf a  = pack_a(*(const bf16x8*)(arow), *(const bf16x8*)(arow + 8));
  v16bf b0 = *(const v16bf*)(b0row);
  v16bf b1 = *(const v16bf*)(b1row);
#pragma unroll 4
  for (int kk = 32; kk < MM; kk += 32) {
    // preload next chunk before consuming current (overlap loads with wmma)
    v16bf an  = pack_a(*(const bf16x8*)(arow + kk), *(const bf16x8*)(arow + kk + 8));
    v16bf b0n = *(const v16bf*)(b0row + kk);
    v16bf b1n = *(const v16bf*)(b1row + kk);
    c0 = wmma_bf16(a, b0, c0);
    c1 = wmma_bf16(a, b1, c1);
    a = an; b0 = b0n; b1 = b1n;
  }
  c0 = wmma_bf16(a, b0, c0);          // epilogue
  c1 = wmma_bf16(a, b1, c1);

  const int rowoff = hi * 8;
#pragma unroll
  for (int r = 0; r < 8; ++r) {
    P[(m0 + rowoff + r) * 256 + n0 + row]      = c0[r];
    P[(m0 + rowoff + r) * 256 + n0 + 16 + row] = c1[r];
  }
}

// ---------------------------------------------------------------------------
// Gather-combine: feats_bf16[i][c] = relu(P[sub][c] + P[obj][128+c] + fc_b[c])
// ---------------------------------------------------------------------------
__global__ void k_combine(const int* __restrict__ hoi, const float* __restrict__ P,
                          const float* __restrict__ fcb, __bf16* __restrict__ feats) {
  int i = blockIdx.x, c = threadIdx.x;
  int s = hoi[3 * i + 1], o = hoi[3 * i + 2];
  float v = P[s * 256 + c] + P[o * 256 + 128 + c] + fcb[c];
  feats[i * DHID + c] = f2bf(fmaxf(v, 0.0f));
}

// ---------------------------------------------------------------------------
// GEMM 2: XG[4096x1024] = feats_bf16[4096x128] @ WihCat^T + biascat
// K = 128 -> 4 fully-unrolled chunks; all loads issue up front.
// ---------------------------------------------------------------------------
__global__ void k_gemm_xg(const __bf16* __restrict__ F, const __bf16* __restrict__ Wih,
                          const float* __restrict__ bias, float* __restrict__ XG) {
  const int wave = blockIdx.x * (blockDim.x >> 5) + (threadIdx.x >> 5);
  const int lane = threadIdx.x & 31;
  const int mtile = wave >> 5;        // 0..255
  const int ng    = wave & 31;        // 0..31 (32 cols each)
  const int m0 = mtile * 16, n0 = ng * 32;
  const int row = lane & 15, hi = lane >> 4;

  const __bf16* arow  = F + (m0 + row) * DHID + hi * 16;
  const __bf16* b0row = Wih + (n0 + row) * DHID + hi * 16;
  const __bf16* b1row = Wih + (n0 + 16 + row) * DHID + hi * 16;

  v16bf av[4], b0v[4], b1v[4];
#pragma unroll
  for (int j = 0; j < 4; ++j) {
    int kk = j * 32;
    av[j]  = pack_a(*(const bf16x8*)(arow + kk), *(const bf16x8*)(arow + kk + 8));
    b0v[j] = *(const v16bf*)(b0row + kk);
    b1v[j] = *(const v16bf*)(b1row + kk);
  }
  v8f c0 = {}; v8f c1 = {};
#pragma unroll
  for (int j = 0; j < 4; ++j) {
    c0 = wmma_bf16(av[j], b0v[j], c0);
    c1 = wmma_bf16(av[j], b1v[j], c1);
  }
  const int rowoff = hi * 8;
#pragma unroll
  for (int r = 0; r < 8; ++r) {
    int orow = m0 + rowoff + r;
    int col0 = n0 + row, col1 = n0 + 16 + row;
    XG[orow * 1024 + col0] = c0[r] + bias[col0];
    XG[orow * 1024 + col1] = c1[r] + bias[col1];
  }
}

// ---------------------------------------------------------------------------
// LSTM recurrence: one persistent workgroup (16 waves) per direction.
// Per step: g[16x512] = XG[t] + h @ w_hh^T via WMMA; gate math in VALU.
// ---------------------------------------------------------------------------
__global__ void k_lstm(const float* __restrict__ XG, const __bf16* __restrict__ whh_all,
                       float* __restrict__ out) {
  const int dir  = blockIdx.x;                 // 0 = forward, 1 = backward
  const int tid  = threadIdx.x;                // 0..511
  const int lane = tid & 31;
  const int wave = tid >> 5;                   // 0..15
  const __bf16* whh = whh_all + dir * G4 * DHID;
  float* rec = out + BIMG * 256;               // rec_repr region of d_out

  __shared__ float  g_lds[16 * G4];            // 32 KB gates
  __shared__ __bf16 h_lds[16 * DHID];          // 4 KB hidden state (bf16)

  float creg[4] = {0.f, 0.f, 0.f, 0.f};        // each thread owns 4 cells
#pragma unroll
  for (int q = 0; q < 4; ++q) h_lds[tid * 4 + q] = f2bf(0.0f);
  __syncthreads();

  const int row = lane & 15, hi = lane >> 4;
  const int n0 = wave * 32;

  for (int step = 0; step < TSEQ; ++step) {
    const int t = dir ? (TSEQ - 1 - step) : step;

    // --- phase 1: g = h @ whh^T (WMMA), + xg, into LDS ---
    v8f c0 = {}; v8f c1 = {};
#pragma unroll
    for (int kk = 0; kk < DHID; kk += 32) {
      const __bf16* pa = &h_lds[row * DHID + kk + hi * 16];
      v16bf a  = pack_a(*(const bf16x8*)pa, *(const bf16x8*)(pa + 8));
      v16bf b0 = *(const v16bf*)(whh + (n0 + row) * DHID + kk + hi * 16);
      v16bf b1 = *(const v16bf*)(whh + (n0 + 16 + row) * DHID + kk + hi * 16);
      c0 = wmma_bf16(a, b0, c0);
      c1 = wmma_bf16(a, b1, c1);
    }
    const int rowoff = hi * 8;
#pragma unroll
    for (int r = 0; r < 8; ++r) {
      int b = rowoff + r;                                    // batch index
      const float* xr = XG + (b * TSEQ + t) * 1024 + dir * G4;
      int col0 = n0 + row, col1 = n0 + 16 + row;
      g_lds[b * G4 + col0] = c0[r] + xr[col0];
      g_lds[b * G4 + col1] = c1[r] + xr[col1];
    }
    __syncthreads();

    // --- phase 2: gate nonlinearities, state update, emit h ---
#pragma unroll
    for (int q = 0; q < 4; ++q) {
      int cell = tid * 4 + q;
      int b = cell >> 7, j = cell & 127;
      float gi = g_lds[b * G4 + j];
      float gf = g_lds[b * G4 + 128 + j];
      float gg = g_lds[b * G4 + 256 + j];
      float go = g_lds[b * G4 + 384 + j];
      float cc = sigm(gf) * creg[q] + sigm(gi) * tanhf(gg);
      creg[q] = cc;
      float h = sigm(go) * tanhf(cc);
      rec[(b * TSEQ + t) * 256 + dir * DHID + j] = h;
      h_lds[b * DHID + j] = f2bf(h);
    }
    __syncthreads();
  }
}

// ---------------------------------------------------------------------------
// context_feats[b][c] = mean over t of rec_repr[b*256+t][c]
// ---------------------------------------------------------------------------
__global__ void k_context(const float* __restrict__ outbuf, float* __restrict__ ctx) {
  int b = blockIdx.x, c = threadIdx.x;
  const float* rec = outbuf + BIMG * 256;
  float s = 0.f;
  for (int t = 0; t < TSEQ; ++t) s += rec[(b * TSEQ + t) * 256 + c];
  ctx[b * 256 + c] = s * (1.0f / (float)TSEQ);
}

// ---------------------------------------------------------------------------
extern "C" void kernel_launch(void* const* d_in, const int* in_sizes, int n_in,
                              void* d_out, int out_size, void* d_ws, size_t ws_size,
                              hipStream_t stream) {
  (void)in_sizes; (void)n_in; (void)out_size; (void)ws_size;
  const float* masks = (const float*)d_in[0];
  const int*   hoi   = (const int*)d_in[2];
  const float* fc_w  = (const float*)d_in[3];
  const float* fc_b  = (const float*)d_in[4];
  const float* wih_f = (const float*)d_in[5];
  const float* whh_f = (const float*)d_in[6];
  const float* bih_f = (const float*)d_in[7];
  const float* bhh_f = (const float*)d_in[8];
  const float* wih_b = (const float*)d_in[9];
  const float* whh_b = (const float*)d_in[10];
  const float* bih_b = (const float*)d_in[11];
  const float* bhh_b = (const float*)d_in[12];

  char* ws = (char*)d_ws;
  __bf16* masks_bf = (__bf16*)(ws + WS_MASKS);
  __bf16* wcat_bf  = (__bf16*)(ws + WS_WCAT);
  float*  P        = (float*)(ws + WS_P);
  __bf16* feats_bf = (__bf16*)(ws + WS_FEATS);
  __bf16* wih_bf   = (__bf16*)(ws + WS_WIH);
  float*  biascat  = (float*)(ws + WS_BIAS);
  __bf16* whh_bf   = (__bf16*)(ws + WS_WHH);
  float*  XG       = (float*)(ws + WS_XG);
  float*  outp     = (float*)d_out;

  // prep: bf16 conversions + weight packing
  k_cvt_bf16<<<2048, 256, 0, stream>>>(masks, masks_bf, NBOX * MM);
  k_build_wcat<<<1024, 256, 0, stream>>>(fc_w, wcat_bf);
  k_cvt_bf16<<<128, 256, 0, stream>>>(whh_f, whh_bf, G4 * DHID);
  k_cvt_bf16<<<128, 256, 0, stream>>>(whh_b, whh_bf + G4 * DHID, G4 * DHID);
  k_build_wih<<<256, 256, 0, stream>>>(wih_f, wih_b, wih_bf);
  k_build_bias<<<4, 256, 0, stream>>>(bih_f, bhh_f, bih_b, bhh_b, biascat);

  // per-box feature GEMM (2048x4096x256, bf16 WMMA)
  k_gemm_box<<<128, 256, 0, stream>>>(masks_bf, wcat_bf, P);
  // gather + relu -> feats
  k_combine<<<NHOI, DHID, 0, stream>>>(hoi, P, fc_b, feats_bf);
  // xg GEMM (4096x128x1024, bf16 WMMA), bias folded
  k_gemm_xg<<<1024, 256, 0, stream>>>(feats_bf, wih_bf, biascat, XG);
  // bidirectional LSTM recurrence (one persistent WG per direction)
  k_lstm<<<2, 512, 0, stream>>>(XG, whh_bf, outp);
  // context means
  k_context<<<BIMG, 256, 0, stream>>>(outp, outp);
}
